// SimpleExpertChoiceMoE_23742579213114
// MI455X (gfx1250) — matrix operand order repaired and checked by reading, not compiled, not run on hardware
//
#include <hip/hip_runtime.h>
#include <hip/hip_bf16.h>
#include <stdint.h>

// Problem constants (B=2, S=2048 -> T=4096; D=1024; E=8; H=4096; top_k=1024)
#define TKN   4096
#define DIM   1024
#define NEXP  8
#define HID   4096
#define KCAP  1024

typedef __bf16 v16bf __attribute__((ext_vector_type(16)));
typedef float  v8f   __attribute__((ext_vector_type(8)));
typedef int    v4i   __attribute__((vector_size(16)));    // matches builtin's V4i

#if defined(__gfx1250__) && __has_builtin(__builtin_amdgcn_global_load_async_to_lds_b128)
#define USE_ASYNC 1
#else
#define USE_ASYNC 0
#endif

#if defined(__gfx1250__) && __has_builtin(__builtin_amdgcn_s_wait_asynccnt)
#define WAIT_ASYNC(n) __builtin_amdgcn_s_wait_asynccnt(n)
#elif defined(__gfx1250__)
#define WAIT_ASYNC(n) asm volatile("s_wait_asynccnt %0" ::"i"(n))
#else
#define WAIT_ASYNC(n)
#endif

#if USE_ASYNC
#define GLB_CAST(p) ((__attribute__((address_space(1))) v4i*)(uintptr_t)(p))
#define LDS_CAST(p) ((__attribute__((address_space(3))) v4i*)(uint32_t)(uintptr_t)(p))
#endif

__device__ __forceinline__ uint16_t f2bf(float f) {
  uint32_t u = __float_as_uint(f);
  u += 0x7FFFu + ((u >> 16) & 1u);   // round-to-nearest-even
  return (uint16_t)(u >> 16);
}

// ---------------- Router matmul + softmax (wave-per-token) ----------------
__global__ __launch_bounds__(256)
void router_kernel(const float* __restrict__ x,
                   const float* __restrict__ rw,
                   float* __restrict__ probsT) {
  const int wid  = (blockIdx.x * blockDim.x + threadIdx.x) >> 5;
  const int lane = threadIdx.x & 31;
  if (wid >= TKN) return;
  const float* xr = x + (size_t)wid * DIM;
  float acc[NEXP];
#pragma unroll
  for (int e = 0; e < NEXP; ++e) acc[e] = 0.0f;
  for (int j = lane; j < DIM; j += 32) {
    const float xv = xr[j];
    const float* r = rw + (size_t)j * NEXP;
#pragma unroll
    for (int e = 0; e < NEXP; ++e) acc[e] = fmaf(xv, r[e], acc[e]);
  }
#pragma unroll
  for (int off = 16; off > 0; off >>= 1) {
#pragma unroll
    for (int e = 0; e < NEXP; ++e) acc[e] += __shfl_xor(acc[e], off, 32);
  }
  if (lane == 0) {
    float mx = acc[0];
#pragma unroll
    for (int e = 1; e < NEXP; ++e) mx = fmaxf(mx, acc[e]);
    float sum = 0.0f;
    float p[NEXP];
#pragma unroll
    for (int e = 0; e < NEXP; ++e) { p[e] = __expf(acc[e] - mx); sum += p[e]; }
    const float inv = 1.0f / sum;
#pragma unroll
    for (int e = 0; e < NEXP; ++e) probsT[(size_t)e * TKN + wid] = p[e] * inv;
  }
}

// ---------------- Per-expert top-K via in-LDS bitonic sort ----------------
__global__ __launch_bounds__(1024)
void topk_kernel(const float* __restrict__ probsT,
                 int* __restrict__ sel,
                 float* __restrict__ wts) {
  const int e = blockIdx.x;
  __shared__ float sv[TKN];
  __shared__ int   si[TKN];
  for (int i = threadIdx.x; i < TKN; i += 1024) {
    sv[i] = probsT[(size_t)e * TKN + i];
    si[i] = i;
  }
  __syncthreads();
  for (int k = 2; k <= TKN; k <<= 1) {
    for (int j = k >> 1; j > 0; j >>= 1) {
      for (int t = threadIdx.x; t < TKN / 2; t += 1024) {
        const int i = ((t & ~(j - 1)) << 1) | (t & (j - 1));
        const int p = i | j;
        const bool desc = ((i & k) == 0);
        const float vi = sv[i], vp = sv[p];
        const bool sw = desc ? (vi < vp) : (vi > vp);
        if (sw) {
          sv[i] = vp; sv[p] = vi;
          const int ti = si[i]; si[i] = si[p]; si[p] = ti;
        }
      }
      __syncthreads();
    }
  }
  for (int t = threadIdx.x; t < KCAP; t += 1024) {
    sel[e * KCAP + t] = si[t];
    wts[e * KCAP + t] = sv[t];
  }
}

// ------- Transposing fp32 -> bf16 convert: in [E][R][C] f32 -> out [E][C][R] bf16 -------
__global__ __launch_bounds__(256)
void cvt_transpose_bf16(const float* __restrict__ in, uint16_t* __restrict__ out,
                        int R, int C) {
  __shared__ uint16_t tile[64][72];                       // 64x64 tile, padded rows
  const int e  = blockIdx.z;
  const int r0 = blockIdx.y * 64;
  const int c0 = blockIdx.x * 64;
  const float* ine   = in  + (size_t)e * R * C;
  uint16_t*    oute  = out + (size_t)e * R * C;
  const int tid = threadIdx.x;
  {
    const int lr = tid >> 2, lc = (tid & 3) * 16;
    const float* src = ine + (size_t)(r0 + lr) * C + c0 + lc;
    union { uint4 q[2]; uint16_t h[16]; } u;
#pragma unroll
    for (int q = 0; q < 4; ++q) {
      const float4 v = ((const float4*)src)[q];
      u.h[4 * q + 0] = f2bf(v.x); u.h[4 * q + 1] = f2bf(v.y);
      u.h[4 * q + 2] = f2bf(v.z); u.h[4 * q + 3] = f2bf(v.w);
    }
    *(uint4*)&tile[lr][lc]     = u.q[0];
    *(uint4*)&tile[lr][lc + 8] = u.q[1];
  }
  __syncthreads();
  {
    const int orow = tid >> 2, ocol = (tid & 3) * 16;
    union { uint4 q[2]; uint16_t h[16]; } u;
#pragma unroll
    for (int i = 0; i < 16; ++i) u.h[i] = tile[ocol + i][orow];
    uint16_t* dst = oute + (size_t)(c0 + orow) * R + r0 + ocol;
    *(uint4*)dst       = u.q[0];
    *(uint4*)(dst + 8) = u.q[1];
  }
}

// ---------------- Gather selected token rows -> bf16 [E,K,D] ----------------
__global__ __launch_bounds__(256)
void gather_bf16(const float* __restrict__ x,
                 const int* __restrict__ sel,
                 uint16_t* __restrict__ xsel) {
  const int row = blockIdx.x;             // e*KCAP + k
  const int s = sel[row];
  const float* src = x + (size_t)s * DIM;
  uint16_t* dst = xsel + (size_t)row * DIM;
  const int i = threadIdx.x;              // 256 threads x 4 elems = 1024
  const float4 v = ((const float4*)src)[i];
  uint64_t p = (uint64_t)f2bf(v.x)
             | ((uint64_t)f2bf(v.y) << 16)
             | ((uint64_t)f2bf(v.z) << 32)
             | ((uint64_t)f2bf(v.w) << 48);
  ((uint64_t*)dst)[i] = p;
}

// ---------------- bf16 WMMA GEMM, per-expert, two epilogues ----------------
// C[e] = A[e](1024 x Kd) * Bt[e](N x Kd)^T ; A row-major [M][Kd], B pre-transposed [N][Kd]
//   epi==0: h = bf16(gelu_exact(C))          (GEMM1: Kd=1024, N=4096)
//   epi==1: atomic scatter wts*C into outf   (GEMM2: Kd=4096, N=1024)
__global__ __launch_bounds__(256)
void moe_gemm(const uint16_t* __restrict__ A, const uint16_t* __restrict__ B,
              int Kd, int N,
              uint16_t* __restrict__ Cbf,
              float* __restrict__ outf,
              const int* __restrict__ sel,
              const float* __restrict__ wts,
              int epi) {
  constexpr int KT = 32, BM = 128, BN = 128, LDT = 40;     // 8-half row pad
  __shared__ uint16_t ldsA[2][BM * LDT];                   // double buffered
  __shared__ uint16_t ldsB[2][BN * LDT];

  const int e  = blockIdx.z;
  const int m0 = blockIdx.y * BM;
  const int n0 = blockIdx.x * BN;
  const uint16_t* Ae = A + (size_t)e * KCAP * Kd;
  const uint16_t* Be = B + (size_t)e * Kd * N;             // [N][Kd] row-major

  const int tid  = threadIdx.x;
  const int w    = tid >> 5, lane = tid & 31;
  const int wm   = w & 3,    wn   = w >> 2;                // 4 x 2 wave grid
  const int lm   = lane & 15, kh  = lane >> 4;

  // tile staging: 128 rows x 32 halfs, 2 threads per row, 16 halfs each
  const int tr = tid >> 1, tc = (tid & 1) * 16;
  const uint16_t* gA = Ae + (size_t)(m0 + tr) * Kd + tc;
  const uint16_t* gB = Be + (size_t)(n0 + tr) * Kd + tc;
  uint16_t* lA = &ldsA[0][tr * LDT + tc];
  uint16_t* lB = &ldsB[0][tr * LDT + tc];
  const int lbufStride = BM * LDT;

  v8f acc[2][4];
#pragma unroll
  for (int mi = 0; mi < 2; ++mi)
#pragma unroll
    for (int f = 0; f < 4; ++f)
#pragma unroll
      for (int r = 0; r < 8; ++r) acc[mi][f][r] = 0.0f;

  auto issue = [&](int s, int buf) {
    const uint16_t* sa = gA + s * KT;
    const uint16_t* sb = gB + s * KT;
    uint16_t* da = lA + buf * lbufStride;
    uint16_t* db = lB + buf * lbufStride;
#if USE_ASYNC
    __builtin_amdgcn_global_load_async_to_lds_b128(GLB_CAST(sa),     LDS_CAST(da),     0, 0);
    __builtin_amdgcn_global_load_async_to_lds_b128(GLB_CAST(sa + 8), LDS_CAST(da + 8), 0, 0);
    __builtin_amdgcn_global_load_async_to_lds_b128(GLB_CAST(sb),     LDS_CAST(db),     0, 0);
    __builtin_amdgcn_global_load_async_to_lds_b128(GLB_CAST(sb + 8), LDS_CAST(db + 8), 0, 0);
#else
    *(uint4*)da       = *(const uint4*)sa;
    *(uint4*)(da + 8) = *(const uint4*)(sa + 8);
    *(uint4*)db       = *(const uint4*)sb;
    *(uint4*)(db + 8) = *(const uint4*)(sb + 8);
#endif
  };

  const int nsteps = Kd / KT;
  issue(0, 0);
  for (int s = 0; s < nsteps; ++s) {
    const int cur = s & 1;
    if (s + 1 < nsteps) {
      issue(s + 1, cur ^ 1);
      WAIT_ASYNC(4);                     // async loads complete in order
    } else {
      WAIT_ASYNC(0);
    }
    __syncthreads();

    union { v16bf v; uint32_t u[8]; } af[2];
#pragma unroll
    for (int mi = 0; mi < 2; ++mi) {
      const int arow = cur * lbufStride + (wm * 32 + mi * 16 + lm) * LDT;
#pragma unroll
      for (int i = 0; i < 8; ++i) {
        const int ka = ((i < 4) ? (2 * i) : (16 + 2 * (i - 4))) + 8 * kh;
        af[mi].u[i] = *(const uint32_t*)&ldsA[0][arow + ka];
      }
    }
#pragma unroll
    for (int f = 0; f < 4; ++f) {
      union { v16bf v; uint32_t u[8]; } bfr;
      const int brow = cur * lbufStride + (wn * 64 + f * 16 + lm) * LDT;
#pragma unroll
      for (int i = 0; i < 8; ++i) {
        const int kb = ((i < 4) ? (2 * i) : (16 + 2 * (i - 4))) + 8 * kh;
        bfr.u[i] = *(const uint32_t*)&ldsB[0][brow + kb];
      }
#pragma unroll
      for (int mi = 0; mi < 2; ++mi) {
        acc[mi][f] = __builtin_amdgcn_wmma_f32_16x16x32_bf16(
            false, af[mi].v, false, bfr.v, (short)0, acc[mi][f], false, false);
      }
    }
    __syncthreads();
  }

  // Epilogue: C VGPR r holds M=r (lanes 0-15) / M=r+8 (lanes 16-31), N=lane&15
#pragma unroll
  for (int mi = 0; mi < 2; ++mi) {
#pragma unroll
    for (int f = 0; f < 4; ++f) {
      const int n = n0 + wn * 64 + f * 16 + lm;
#pragma unroll
      for (int r = 0; r < 8; ++r) {
        const int m = m0 + wm * 32 + mi * 16 + r + 8 * kh;
        const float v = acc[mi][f][r];
        if (epi == 0) {
          const float g = 0.5f * v * (1.0f + erff(v * 0.70710678118654752f));
          Cbf[(size_t)e * KCAP * N + (size_t)m * N + n] = f2bf(g);
        } else {
          const int tok  = sel[e * KCAP + m];
          const float wg = wts[e * KCAP + m];
          unsafeAtomicAdd(&outf[(size_t)tok * DIM + n], wg * v);
        }
      }
    }
  }
}

// ---------------------------------------------------------------------------
extern "C" void kernel_launch(void* const* d_in, const int* in_sizes, int n_in,
                              void* d_out, int out_size, void* d_ws, size_t ws_size,
                              hipStream_t stream) {
  const float* x  = (const float*)d_in[0];
  const float* rw = (const float*)d_in[1];
  const float* w1 = (const float*)d_in[2];
  const float* w2 = (const float*)d_in[3];
  float* out = (float*)d_out;

  char* ws = (char*)d_ws;
  size_t off = 0;
  float* probsT = (float*)(ws + off);   off += (size_t)NEXP * TKN * 4;
  int*   sel    = (int*)(ws + off);     off += (size_t)NEXP * KCAP * 4;
  float* wts    = (float*)(ws + off);   off += (size_t)NEXP * KCAP * 4;
  off = (off + 255) & ~(size_t)255;
  uint16_t* xsel = (uint16_t*)(ws + off); off += (size_t)NEXP * KCAP * DIM * 2; // 16.8 MB
  uint16_t* w1t  = (uint16_t*)(ws + off); off += (size_t)NEXP * DIM * HID * 2;  // 67 MB [E][H][D]
  uint16_t* w2t  = (uint16_t*)(ws + off); off += (size_t)NEXP * HID * DIM * 2;  // 67 MB [E][D][H]
  uint16_t* hbuf = (uint16_t*)(ws + off); off += (size_t)NEXP * KCAP * HID * 2; // 67 MB

  (void)hipMemsetAsync(out, 0, (size_t)TKN * DIM * sizeof(float), stream);

  router_kernel<<<TKN * 32 / 256, 256, 0, stream>>>(x, rw, probsT);
  topk_kernel<<<NEXP, 1024, 0, stream>>>(probsT, sel, wts);

  // w1 [E][D][H] -> w1t [E][H][D]; w2 [E][H][D] -> w2t [E][D][H]
  cvt_transpose_bf16<<<dim3(HID / 64, DIM / 64, NEXP), 256, 0, stream>>>(w1, w1t, DIM, HID);
  cvt_transpose_bf16<<<dim3(DIM / 64, HID / 64, NEXP), 256, 0, stream>>>(w2, w2t, HID, DIM);

  gather_bf16<<<NEXP * KCAP, 256, 0, stream>>>(x, sel, xsel);

  // GEMM1: [1024 x 1024] x [1024 x 4096] per expert, GELU -> bf16 h
  moe_gemm<<<dim3(HID / 128, KCAP / 128, NEXP), 256, 0, stream>>>(
      xsel, w1t, DIM, HID, hbuf, nullptr, nullptr, nullptr, 0);
  // GEMM2: [1024 x 4096] x [4096 x 1024] per expert, weighted atomic scatter
  moe_gemm<<<dim3(DIM / 128, KCAP / 128, NEXP), 256, 0, stream>>>(
      hbuf, w2t, HID, DIM, nullptr, out, sel, wts, 1);
}
